// RNN_model_GloVe_28750511079514
// MI455X (gfx1250) — matrix operand order repaired and checked by reading, not compiled, not run on hardware
//
#include <hip/hip_runtime.h>
#include <hip/hip_bf16.h>
#include <math.h>

typedef __attribute__((ext_vector_type(16))) __bf16 v16bf;
typedef __attribute__((ext_vector_type(8)))  float  v8f;
typedef unsigned short u16;
typedef unsigned int   u32;

#define B_   64
#define T_   512
#define D_   300
#define H_   1024
#define KX   320      // D padded to multiple of 32
#define NWG  64
#define CHW  16       // channels per workgroup (H_/NWG)
#define TPB  256
#define EPSV 1e-5f

// ---- ws layout (bytes) ----
#define OFF_WHH  0u                    // bf16 [4096][1024]   8,388,608
#define OFF_WIH  8388608u              // bf16 [4096][320]    2,621,440
#define OFF_XB   11010048u             // bf16 [512][64][320] 20,971,520
#define OFF_H    31981568u             // bf16 [2][64][1024]    262,144
#define OFF_POOL 32243712u             // f32  [64][1024]       262,144
#define OFF_CNT  32505856u             // u32 counter

// ---- LDS layout (bytes) ----
#define L_WHH 0          // u16 [64][1024] 131072
#define L_WIH 131072     // u16 [64][320]   40960
#define L_GB  172032     // f32 [64][64]    16384
#define L_HB  188416     // f32 [64][16]     4096
#define L_CB  192512     // f32 [64][16]     4096
#define L_PB  196608     // f32 [64][16]     4096
#define L_MU  200704     // f32 [16]
#define L_RS  200768     // f32 [16]
#define LDS_BYTES 200832

union AFrag { v16bf v; uint4 q[2]; };

__device__ __forceinline__ u16 f2bf(float f) {           // round-to-nearest-even
    u32 u = __float_as_uint(f);
    u32 r = u + 0x7FFFu + ((u >> 16) & 1u);
    return (u16)(r >> 16);
}
__device__ __forceinline__ float sigf(float x) { return 1.f / (1.f + expf(-x)); }

// ---------------- Kernel 1: pack / convert / init ----------------
__global__ void pack_kernel(const float* __restrict__ x,
                            const float* __restrict__ W_ih,
                            const float* __restrict__ W_hh,
                            u16* __restrict__ whh_b, u16* __restrict__ wih_b,
                            u16* __restrict__ xb, u16* __restrict__ hbuf,
                            u32* __restrict__ cnt) {
    const long gid = (long)blockIdx.x * blockDim.x + threadIdx.x;
    const long str = (long)gridDim.x * blockDim.x;
    for (long i = gid; i < 4096L * 1024L; i += str) whh_b[i] = f2bf(W_hh[i]);
    for (long i = gid; i < 4096L * KX; i += str) {
        long col = i / KX, k = i % KX;
        wih_b[i] = (k < D_) ? f2bf(W_ih[col * D_ + k]) : (u16)0;
    }
    for (long i = gid; i < (long)T_ * B_ * KX; i += str) {
        long t = i / (B_ * KX), r = i % (B_ * KX), b = r / KX, k = r % KX;
        xb[i] = (k < D_) ? f2bf(x[(b * T_ + t) * D_ + k]) : (u16)0;
    }
    for (long i = gid; i < B_ * H_; i += str) hbuf[i] = 0;   // h0 = 0 (ping buffer)
    if (gid == 0) *cnt = 0u;
}

// ---------------- Kernel 2: persistent LSTM scan ----------------
__global__ __launch_bounds__(TPB) void lstm_persistent(
    const u16* __restrict__ xb, const u16* __restrict__ whh_b,
    const u16* __restrict__ wih_b,
    const float* __restrict__ b_ih, const float* __restrict__ b_hh,
    const float* __restrict__ gamma, const float* __restrict__ beta,
    const float* __restrict__ mask,
    u16* __restrict__ hbuf, float* __restrict__ pooled,
    u32* __restrict__ cnt) {
    extern __shared__ char smem[];
    u16*   WH = (u16*)(smem + L_WHH);
    u16*   WI = (u16*)(smem + L_WIH);
    float* GB = (float*)(smem + L_GB);
    float* HB = (float*)(smem + L_HB);
    float* CB = (float*)(smem + L_CB);
    float* PB = (float*)(smem + L_PB);
    float* MU = (float*)(smem + L_MU);
    float* RS = (float*)(smem + L_RS);

    const int tid  = threadIdx.x;
    const int wg   = blockIdx.x;          // owns channels [wg*16, wg*16+16)
    const int wave = tid >> 5;
    const int lane = tid & 31;
    const int m    = wave & 3;            // batch row-tile
    const int g0t  = (wave >> 2) * 2;     // gate types handled by this wave
    const int g1t  = g0t + 1;
    const int laneN = lane & 15;
    const int hi    = lane >> 4;
    const int rowA  = m * 16 + laneN;
    const int koffA = hi ? 16 : 0;        // A: contiguous 16 K at koffA
    const int koffB = hi ? 16 : 0;        // B: contiguous 16 K at koffB

    // ---- stage weight slices into LDS (reused for all 512 steps) ----
    for (int u = tid; u < 64 * 128; u += TPB) {          // W_hh: 64 cols x 128 x (8 bf16)
        int c = u >> 7, kc = u & 127;
        int colg = (c >> 4) * H_ + wg * CHW + (c & 15);
        *(uint4*)(WH + c * 1024 + kc * 8) =
            *(const uint4*)(whh_b + (size_t)colg * 1024 + kc * 8);
    }
    for (int u = tid; u < 64 * 40; u += TPB) {           // W_ih: 64 cols x 40 x (8 bf16)
        int c = u / 40, kc = u % 40;
        int colg = (c >> 4) * H_ + wg * CHW + (c & 15);
        *(uint4*)(WI + c * KX + kc * 8) =
            *(const uint4*)(wih_b + (size_t)colg * KX + kc * 8);
    }
    for (int e = tid; e < B_ * CHW; e += TPB) { CB[e] = 0.f; PB[e] = -3.4e38f; }
    __syncthreads();

    for (int s = 0; s < T_; ++s) {
        v8f acc0 = {}; v8f acc1 = {};
        // -------- input contribution: x_t[64xKX] @ W_ih_slice --------
        const u16* xt = xb + (size_t)s * (B_ * KX);
        for (int k0 = 0; k0 < KX; k0 += 32) {
            AFrag a, b0, b1;
            const u16* ar = xt + (size_t)rowA * KX + k0;
            a.q[0] = *(const uint4*)(ar + koffA);
            a.q[1] = *(const uint4*)(ar + koffA + 8);
            const u16* br0 = WI + (g0t * 16 + laneN) * KX + k0;
            const u16* br1 = WI + (g1t * 16 + laneN) * KX + k0;
            b0.q[0] = *(const uint4*)(br0 + koffB);
            b0.q[1] = *(const uint4*)(br0 + koffB + 8);
            b1.q[0] = *(const uint4*)(br1 + koffB);
            b1.q[1] = *(const uint4*)(br1 + koffB + 8);
            acc0 = __builtin_amdgcn_wmma_f32_16x16x32_bf16(false, a.v, false, b0.v,
                                                           (short)0, acc0, false, false);
            acc1 = __builtin_amdgcn_wmma_f32_16x16x32_bf16(false, a.v, false, b1.v,
                                                           (short)0, acc1, false, false);
        }
        // -------- recurrent contribution: h[64x1024] @ W_hh_slice --------
        const u16* hp = hbuf + (size_t)(s & 1) * (B_ * H_);
        for (int k0 = 0; k0 < H_; k0 += 32) {
            AFrag a, b0, b1;
            const u16* ar = hp + (size_t)rowA * H_ + k0;
            a.q[0] = *(const uint4*)(ar + koffA);
            a.q[1] = *(const uint4*)(ar + koffA + 8);
            const u16* br0 = WH + (g0t * 16 + laneN) * 1024 + k0;
            const u16* br1 = WH + (g1t * 16 + laneN) * 1024 + k0;
            b0.q[0] = *(const uint4*)(br0 + koffB);
            b0.q[1] = *(const uint4*)(br0 + koffB + 8);
            b1.q[0] = *(const uint4*)(br1 + koffB);
            b1.q[1] = *(const uint4*)(br1 + koffB + 8);
            acc0 = __builtin_amdgcn_wmma_f32_16x16x32_bf16(false, a.v, false, b0.v,
                                                           (short)0, acc0, false, false);
            acc1 = __builtin_amdgcn_wmma_f32_16x16x32_bf16(false, a.v, false, b1.v,
                                                           (short)0, acc1, false, false);
        }
        // -------- scatter gate tiles to LDS (C/D layout) --------
        for (int r = 0; r < 8; ++r) {
            int gr = m * 16 + r + hi * 8;
            GB[gr * 64 + g0t * 16 + laneN] = acc0[r];
            GB[gr * 64 + g1t * 16 + laneN] = acc1[r];
        }
        __syncthreads();

        // -------- elementwise LSTM cell (torch gate order i,f,g,o) --------
        u16* hn_g = hbuf + (size_t)((s + 1) & 1) * (B_ * H_);
        for (int e = tid; e < B_ * CHW; e += TPB) {
            int row = e >> 4, ch = e & 15, cg = wg * CHW + ch;
            float gi = GB[row * 64 +  0 + ch] + b_ih[0 * H_ + cg] + b_hh[0 * H_ + cg];
            float gf = GB[row * 64 + 16 + ch] + b_ih[1 * H_ + cg] + b_hh[1 * H_ + cg];
            float gg = GB[row * 64 + 32 + ch] + b_ih[2 * H_ + cg] + b_hh[2 * H_ + cg];
            float go = GB[row * 64 + 48 + ch] + b_ih[3 * H_ + cg] + b_hh[3 * H_ + cg];
            float cp = CB[e];
            float c2 = sigf(gf) * cp + sigf(gi) * tanhf(gg);
            float h2 = sigf(go) * tanhf(c2);
            CB[e] = c2; HB[e] = h2;
            hn_g[row * H_ + cg] = f2bf(h2);       // broadcast raw h for next step
        }
        __syncthreads();

        // -------- BatchNorm stats over batch (biased) --------
        if (tid < CHW) {
            float su = 0.f, sq = 0.f;
            for (int r = 0; r < B_; ++r) { float v = HB[r * CHW + tid]; su += v; sq += v * v; }
            float mu = su * (1.f / B_);
            float var = sq * (1.f / B_) - mu * mu;
            MU[tid] = mu; RS[tid] = rsqrtf(var + EPSV);
        }
        __syncthreads();

        // -------- normalize + locked dropout + running max --------
        for (int e = tid; e < B_ * CHW; e += TPB) {
            int row = e >> 4, ch = e & 15, cg = wg * CHW + ch;
            float hn = gamma[cg] * (HB[e] - MU[ch]) * RS[ch] + beta[cg];
            float o  = hn * mask[row * H_ + cg];
            PB[e] = fmaxf(PB[e], o);
        }
        __syncthreads();

        // -------- device-wide barrier (monotonic counter) --------
        if (tid == 0) {
            __threadfence();
            __hip_atomic_fetch_add(cnt, 1u, __ATOMIC_RELEASE, __HIP_MEMORY_SCOPE_AGENT);
            u32 target = (u32)(s + 1) * NWG;
            while (__hip_atomic_load(cnt, __ATOMIC_ACQUIRE, __HIP_MEMORY_SCOPE_AGENT) < target)
                __builtin_amdgcn_s_sleep(2);
        }
        __syncthreads();
    }

    for (int e = tid; e < B_ * CHW; e += TPB) {
        int row = e >> 4, ch = e & 15;
        pooled[row * H_ + wg * CHW + ch] = PB[e];
    }
}

// ---------------- Kernel 3: FC head + BCE loss ----------------
__global__ void head_kernel(const float* __restrict__ pooled,
                            const float* __restrict__ fc_w,
                            const float* __restrict__ fc_b,
                            const float* __restrict__ tlab,
                            float* __restrict__ out) {
    __shared__ float zs[B_];
    int tid = threadIdx.x;
    if (tid < B_) {
        float acc = fc_b[0];
        for (int h = 0; h < H_; ++h) acc += pooled[tid * H_ + h] * fc_w[h];
        zs[tid] = acc;
        out[1 + tid] = acc;                       // z logits
    }
    __syncthreads();
    if (tid == 0) {
        float loss = 0.f;
        for (int b = 0; b < B_; ++b) {
            float z = zs[b];
            loss += fmaxf(z, 0.f) + log1pf(expf(-fabsf(z))) - z * tlab[b];
        }
        out[0] = loss * (1.f / B_);               // mean BCEWithLogits
    }
}

extern "C" void kernel_launch(void* const* d_in, const int* in_sizes, int n_in,
                              void* d_out, int out_size, void* d_ws, size_t ws_size,
                              hipStream_t stream) {
    (void)in_sizes; (void)n_in; (void)out_size; (void)ws_size;
    const float* x     = (const float*)d_in[0];
    const float* tlab  = (const float*)d_in[1];
    const float* mask  = (const float*)d_in[2];
    const float* W_ih  = (const float*)d_in[3];
    const float* W_hh  = (const float*)d_in[4];
    const float* b_ih  = (const float*)d_in[5];
    const float* b_hh  = (const float*)d_in[6];
    const float* gamma = (const float*)d_in[7];
    const float* beta  = (const float*)d_in[8];
    const float* fc_w  = (const float*)d_in[9];
    const float* fc_b  = (const float*)d_in[10];

    char* ws = (char*)d_ws;
    u16* whh_b  = (u16*)(ws + OFF_WHH);
    u16* wih_b  = (u16*)(ws + OFF_WIH);
    u16* xb     = (u16*)(ws + OFF_XB);
    u16* hbuf   = (u16*)(ws + OFF_H);
    float* pool = (float*)(ws + OFF_POOL);
    u32* cnt    = (u32*)(ws + OFF_CNT);
    float* out  = (float*)d_out;

    pack_kernel<<<4096, TPB, 0, stream>>>(x, W_ih, W_hh, whh_b, wih_b, xb, hbuf, cnt);
    lstm_persistent<<<NWG, TPB, LDS_BYTES, stream>>>(xb, whh_b, wih_b, b_ih, b_hh,
                                                     gamma, beta, mask, hbuf, pool, cnt);
    head_kernel<<<1, 64, 0, stream>>>(pool, fc_w, fc_b, tlab, out);
}
